// MambaDecoderPatchExpandingBlock_64630667870744
// MI455X (gfx1250) — matrix operand order repaired and checked by reading, not compiled, not run on hardware
//
#include <hip/hip_runtime.h>
#include <hip/hip_bf16.h>

// ---------------- WMMA types (CDNA5 / gfx1250, wave32) ----------------
typedef __attribute__((ext_vector_type(16))) __bf16 v16bf;
typedef __attribute__((ext_vector_type(4)))  __bf16 v4bf;
typedef __attribute__((ext_vector_type(8)))  float  v8f;
typedef __attribute__((ext_vector_type(4)))  float  v4f;

// ---------------- Constants from the reference ----------------
#define LSEQ   4097          // 1 + 4*1024 tokens per batch
#define MROWS  8194          // 2 * LSEQ
#define DMODEL 256
#define DINNER 512
#define DSTATE 16

#define LDSP 40              // LDS row pitch in bf16: 80B = 5*16B, conflict-free

__device__ __forceinline__ v4bf cvt4(v4f v) {
  v4bf b = {(__bf16)v.x, (__bf16)v.y, (__bf16)v.z, (__bf16)v.w};
  return b;
}

// =====================================================================
// GEMM tile body: C[M,N] = A[M,K] * W[N,K]^T  (weights out-major, K-minor).
// bf16 WMMA, f32 accumulate.  Macro tile 64(M) x 64(N): 8 waves as
// 4(M) x 2(N); each wave computes two 16x16 tiles (n, n+32) sharing one A
// fragment -> 2 v_wmma per K-chunk.  FULL=true: interior tile, no guards.
// =====================================================================
template <bool FULL>
__device__ __forceinline__ void gemm_tile(
    const float* __restrict__ A, const float* __restrict__ W,
    float* __restrict__ C, int M, int N, int K,
    int m0, int n0, __bf16* As, __bf16* Bs) {
  const int tid  = threadIdx.x;
  const int wid  = tid >> 5;
  const int lane = tid & 31;
  const int wm  = (wid & 3) * 16;    // wave's M slice (0..48)
  const int wn0 = (wid >> 2) * 16;   // wave's first N slice (0 or 16)
  const int lm = lane & 15;
  const int hi = lane >> 4;

  // Staging geometry: 64x32 tile = 512 float4 units; 256 threads x 2 units.
  const int r0 = tid >> 3;           // rows 0..31
  const int r1 = r0 + 32;            // rows 32..63
  const int c4 = (tid & 7) * 4;      // 0..28

  const bool pa0 = FULL || (m0 + r0 < M);
  const bool pa1 = FULL || (m0 + r1 < M);
  const bool pb0 = FULL || (n0 + r0 < N);
  const bool pb1 = FULL || (n0 + r1 < N);
  const float* a0 = A + (size_t)(m0 + r0) * K + c4;
  const float* a1 = A + (size_t)(m0 + r1) * K + c4;
  const float* b0 = W + (size_t)(n0 + r0) * K + c4;
  const float* b1 = W + (size_t)(n0 + r1) * K + c4;

  v8f acc0 = {};
  v8f acc1 = {};
  for (int k0 = 0; k0 < K; k0 += 32) {
    // Issue all four b128 loads, then convert+store (single loadcnt wait).
    v4f va0 = {0.f, 0.f, 0.f, 0.f}, va1 = va0, vb0 = va0, vb1 = va0;
    if (pa0) va0 = *(const v4f*)(a0 + k0);
    if (pa1) va1 = *(const v4f*)(a1 + k0);
    if (pb0) vb0 = *(const v4f*)(b0 + k0);
    if (pb1) vb1 = *(const v4f*)(b1 + k0);
    if (k0 + 32 < K) {
      if (pa0) __builtin_prefetch(a0 + k0 + 32, 0, 1);
      if (pb0) __builtin_prefetch(b0 + k0 + 32, 0, 1);
    }
    *(v4bf*)(&As[r0 * LDSP + c4]) = cvt4(va0);
    *(v4bf*)(&As[r1 * LDSP + c4]) = cvt4(va1);
    *(v4bf*)(&Bs[r0 * LDSP + c4]) = cvt4(vb0);
    *(v4bf*)(&Bs[r1 * LDSP + c4]) = cvt4(vb1);
    __syncthreads();

    // Gather fragments per CDNA5 16-bit WMMA VGPR layouts
    // (coalesces to ds_load_b128 pairs; LDSP pitch avoids bank conflicts).
    union { v16bf v; unsigned u[8]; } afrag, bfrag0, bfrag1;
#pragma unroll
    for (int vv = 0; vv < 8; ++vv) {
      const int ka = (vv >> 2) * 16 + hi * 8 + (vv & 3) * 2;  // A: 16x32
      afrag.u[vv] = *(const unsigned*)&As[(wm + lm) * LDSP + ka];
      const int kb = (vv >> 2) * 16 + hi * 8 + (vv & 3) * 2;  // B: 32x16
      bfrag0.u[vv] = *(const unsigned*)&Bs[(wn0 + lm) * LDSP + kb];
      bfrag1.u[vv] = *(const unsigned*)&Bs[(wn0 + 32 + lm) * LDSP + kb];
    }
    acc0 = __builtin_amdgcn_wmma_f32_16x16x32_bf16(
        false, afrag.v, false, bfrag0.v, (short)0, acc0, false, false);
    acc1 = __builtin_amdgcn_wmma_f32_16x16x32_bf16(
        false, afrag.v, false, bfrag1.v, (short)0, acc1, false, false);
    __syncthreads();
  }
  // C layout: VGPR r holds M = r + 8*hi, N = lane&15.
#pragma unroll
  for (int r = 0; r < 8; ++r) {
    int gm = m0 + wm + r + 8 * hi;
    int gn = n0 + wn0 + lm;
    if (FULL || gm < M) {
      if (FULL || gn < N)      C[(size_t)gm * N + gn]      = acc0[r];
      if (FULL || gn + 32 < N) C[(size_t)gm * N + gn + 32] = acc1[r];
    }
  }
}

__global__ __launch_bounds__(256) void gemm_bf16_wmma(
    const float* __restrict__ A, const float* __restrict__ W,
    float* __restrict__ C, int M, int N, int K) {
  __shared__ __attribute__((aligned(16))) __bf16 As[64 * LDSP];
  __shared__ __attribute__((aligned(16))) __bf16 Bs[64 * LDSP];
  const int m0 = blockIdx.y * 64;
  const int n0 = blockIdx.x * 64;
  if (m0 + 64 <= M && n0 + 64 <= N)
    gemm_tile<true>(A, W, C, M, N, K, m0, n0, As, Bs);
  else
    gemm_tile<false>(A, W, C, M, N, K, m0, n0, As, Bs);
}

// =====================================================================
// Pack toks: x[:,1:,:] (2,1024,512) -> contiguous (2048,512)
// =====================================================================
__global__ void pack_toks_kernel(const float* __restrict__ x, float* __restrict__ p) {
  int idx = blockIdx.x * blockDim.x + threadIdx.x;
  if (idx >= 2048 * 512) return;
  int k = idx & 511;
  int i = idx >> 9;            // 0..2047
  int b = i >> 10;
  int j = i & 1023;
  p[idx] = x[((size_t)(b * 1025 + 1 + j)) * 512 + k];
}

// =====================================================================
// Pixel-shuffle + LayerNorm(256) + skip, writing rows 1.. of xx
// =====================================================================
__global__ __launch_bounds__(256) void ln_shuffle_kernel(
    const float* __restrict__ e, const float* __restrict__ skip,
    const float* __restrict__ g, const float* __restrict__ bta,
    float* __restrict__ xx) {
  int blk = blockIdx.x;            // 0..8191
  int m = threadIdx.x;
  int b = blk >> 12;
  int rem = blk & 4095;
  int t = rem >> 10;
  int r = rem & 1023;
  int q = r & 1, w = (r >> 1) & 15, p = (r >> 5) & 1, h = r >> 6;
  int n  = h * 16 + w;
  int ch = (p * 2 + q) * 256 + m;
  float val = e[((size_t)((b * 4 + t) * 256 + n)) * 1024 + ch];

  float s = val, s2 = val * val;
#pragma unroll
  for (int o = 16; o; o >>= 1) { s += __shfl_xor(s, o, 32); s2 += __shfl_xor(s2, o, 32); }
  __shared__ float ps[8], ps2[8];
  int wid = m >> 5, lane = m & 31;
  if (lane == 0) { ps[wid] = s; ps2[wid] = s2; }
  __syncthreads();
  if (wid == 0) {
    float a  = (lane < 8) ? ps[lane]  : 0.f;
    float a2 = (lane < 8) ? ps2[lane] : 0.f;
#pragma unroll
    for (int o = 4; o; o >>= 1) { a += __shfl_xor(a, o, 32); a2 += __shfl_xor(a2, o, 32); }
    if (lane == 0) { ps[0] = a; ps2[0] = a2; }
  }
  __syncthreads();
  float mean = ps[0] * (1.0f / 256.0f);
  float var  = ps2[0] * (1.0f / 256.0f) - mean * mean;
  float nv = (val - mean) * rsqrtf(var + 1e-5f) * g[m] + bta[m];
  int row = b * LSEQ + 1 + t * 1024 + r;
  xx[(size_t)row * 256 + m] = nv + skip[(size_t)row * 256 + m];
}

// cls token: (2,1,512)@(256,512)^T + bias + skip -> row 0 of each batch
__global__ __launch_bounds__(256) void cls_kernel(
    const float* __restrict__ x, const float* __restrict__ wc,
    const float* __restrict__ bc, const float* __restrict__ skip,
    float* __restrict__ xx) {
  int b = blockIdx.x, m = threadIdx.x;
  const float* xr = x + (size_t)b * 1025 * 512;
  float s = bc[m];
  for (int k = 0; k < 512; ++k) s += xr[k] * wc[m * 512 + k];
  int row = b * LSEQ;
  xx[(size_t)row * 256 + m] = s + skip[(size_t)row * 256 + m];
}

// =====================================================================
// Depthwise causal conv(4) + bias + SiLU on the xi half of xz
// =====================================================================
__global__ void conv_silu_kernel(const float* __restrict__ xz,
                                 const float* __restrict__ cw,
                                 const float* __restrict__ cb,
                                 float* __restrict__ xi) {
  int idx = blockIdx.x * blockDim.x + threadIdx.x;
  if (idx >= MROWS * DINNER) return;
  int d = idx & 511;
  int row = idx >> 9;                 // b*LSEQ + l
  int b = row / LSEQ;
  int l = row - b * LSEQ;
  float s = cb[d];
#pragma unroll
  for (int k = 0; k < 4; ++k) {
    int ls = l - 3 + k;
    if (ls >= 0) s += cw[d * 4 + k] * xz[((size_t)(b * LSEQ + ls)) * 1024 + d];
  }
  float sig = 1.0f / (1.0f + __expf(-s));
  xi[idx] = s * sig;
}

// delta = softplus(dt @ Wdt^T + bdt)   (K=16, pure VALU)
__global__ void delta_kernel(const float* __restrict__ dbc,
                             const float* __restrict__ wdt,
                             const float* __restrict__ bdt,
                             float* __restrict__ delta) {
  int idx = blockIdx.x * blockDim.x + threadIdx.x;
  if (idx >= MROWS * DINNER) return;
  int d = idx & 511;
  int row = idx >> 9;
  float s = bdt[d];
#pragma unroll
  for (int r = 0; r < 16; ++r) s += dbc[(size_t)row * 48 + r] * wdt[d * 16 + r];
  delta[idx] = (s > 20.0f) ? s : log1pf(__expf(s));
}

// =====================================================================
// Selective scan: n=0..15 across half-wave lanes; one (b,d) per 16 lanes.
// Fuses D-skip and SiLU(z) gate into the lane-0 write.
// =====================================================================
__global__ __launch_bounds__(128) void scan_kernel(
    const float* __restrict__ dbc, const float* __restrict__ delta,
    const float* __restrict__ xz,  const float* __restrict__ xi,
    const float* __restrict__ alog, const float* __restrict__ Dp,
    float* __restrict__ y) {
  int g = blockIdx.x * (blockDim.x >> 4) + (threadIdx.x >> 4);  // 0..1023
  int n = threadIdx.x & 15;
  int b = g >> 9;
  int d = g & 511;
  float a  = -__expf(alog[d * 16 + n]);
  float Dd = Dp[d];
  float h = 0.0f;
  int base = b * LSEQ;
  for (int l = 0; l < LSEQ; ++l) {
    size_t row = (size_t)(base + l);
    float dlt = delta[row * 512 + d];
    float Bn  = dbc[row * 48 + 16 + n];
    float Cn  = dbc[row * 48 + 32 + n];
    float xv  = xi[row * 512 + d];
    h = __fmaf_rn(h, __expf(dlt * a), dlt * Bn * xv);
    float yp = h * Cn;
    yp += __shfl_xor(yp, 8, 32);
    yp += __shfl_xor(yp, 4, 32);
    yp += __shfl_xor(yp, 2, 32);
    yp += __shfl_xor(yp, 1, 32);
    if (n == 0) {
      float z  = xz[row * 1024 + 512 + d];
      float sz = z / (1.0f + __expf(-z));
      y[row * 512 + d] = (yp + xv * Dd) * sz;
    }
  }
}

// =====================================================================
// BatchNorm over (B,L): per-feature mean/var, then apply to d_out
// =====================================================================
__global__ __launch_bounds__(256) void bn_stats_kernel(const float* __restrict__ xx,
                                                       float* __restrict__ stats) {
  int f = blockIdx.x;
  float s = 0.f, s2 = 0.f;
  for (int row = threadIdx.x; row < MROWS; row += 256) {
    float v = xx[(size_t)row * 256 + f];
    s += v; s2 += v * v;
  }
#pragma unroll
  for (int o = 16; o; o >>= 1) { s += __shfl_xor(s, o, 32); s2 += __shfl_xor(s2, o, 32); }
  __shared__ float ps[8], ps2[8];
  int wid = threadIdx.x >> 5, lane = threadIdx.x & 31;
  if (lane == 0) { ps[wid] = s; ps2[wid] = s2; }
  __syncthreads();
  if (threadIdx.x == 0) {
    float a = 0.f, a2 = 0.f;
    for (int i = 0; i < 8; ++i) { a += ps[i]; a2 += ps2[i]; }
    float mean = a / (float)MROWS;
    stats[f] = mean;
    stats[256 + f] = a2 / (float)MROWS - mean * mean;
  }
}

__global__ void bn_apply_kernel(const float* __restrict__ xx,
                                const float* __restrict__ stats,
                                const float* __restrict__ g,
                                const float* __restrict__ bta,
                                float* __restrict__ out, int total) {
  int i = blockIdx.x * blockDim.x + threadIdx.x;
  if (i >= total) return;
  int f = i & 255;
  out[i] = (xx[i] - stats[f]) * rsqrtf(stats[256 + f] + 1e-5f) * g[f] + bta[f];
}

__global__ void tail_kernel(float* __restrict__ out, int out_size) {
  if (blockIdx.x == 0 && threadIdx.x == 0 && out_size > MROWS * DMODEL)
    out[MROWS * DMODEL] = 1024.0f;   // second tuple element: 4*N
}

// =====================================================================
// Host side
// =====================================================================
extern "C" void kernel_launch(void* const* d_in, const int* in_sizes, int n_in,
                              void* d_out, int out_size, void* d_ws, size_t ws_size,
                              hipStream_t stream) {
  (void)in_sizes; (void)n_in; (void)ws_size;
  const float* x    = (const float*)d_in[0];
  const float* skip = (const float*)d_in[1];
  const float* expw = (const float*)d_in[2];
  const float* lng  = (const float*)d_in[3];
  const float* lnb  = (const float*)d_in[4];
  const float* clsw = (const float*)d_in[5];
  const float* clsb = (const float*)d_in[6];
  const float* wiAll = (const float*)d_in[7];   // (2,1024,256)
  const float* cwAll = (const float*)d_in[8];   // (2,512,4)
  const float* cbAll = (const float*)d_in[9];   // (2,512)
  const float* wxAll = (const float*)d_in[10];  // (2,48,512)
  const float* wdtAll = (const float*)d_in[11]; // (2,512,16)
  const float* bdtAll = (const float*)d_in[12]; // (2,512)
  const float* alogAll = (const float*)d_in[13];// (2,512,16)
  const float* dpAll = (const float*)d_in[14];  // (2,512)
  const float* woAll = (const float*)d_in[15];  // (2,256,512)
  const float* bng = (const float*)d_in[16];
  const float* bnb = (const float*)d_in[17];

  float* ws = (float*)d_ws;
  size_t off = 0;
  float* xxA  = ws + off; off += (size_t)MROWS * DMODEL;
  float* xxB  = ws + off; off += (size_t)MROWS * DMODEL;
  float* xz   = ws + off; off += (size_t)MROWS * 1024;
  float* xi   = ws + off; off += (size_t)MROWS * DINNER;
  float* dbc  = ws + off; off += (size_t)MROWS * 48;
  float* dlt  = ws + off; off += (size_t)MROWS * DINNER;
  float* yb   = ws + off; off += (size_t)MROWS * DINNER;
  float* stats = ws + off; off += 512;
  float* pack = yb;   // alias: 2048*512 fits in y region, consumed before y is produced
  float* ebuf = xz;   // alias: 2048*1024 fits in xz region, consumed before xz is produced

  const int TB = 256;
  auto gemm = [&](const float* A, const float* W, float* C, int M, int N, int K) {
    dim3 grid((N + 63) / 64, (M + 63) / 64);
    gemm_bf16_wmma<<<grid, TB, 0, stream>>>(A, W, C, M, N, K);
  };

  // ---- patch-expand path ----
  pack_toks_kernel<<<(2048 * 512 + TB - 1) / TB, TB, 0, stream>>>(x, pack);
  gemm(pack, expw, ebuf, 2048, 1024, 512);
  ln_shuffle_kernel<<<8192, TB, 0, stream>>>(ebuf, skip, lng, lnb, xxA);
  cls_kernel<<<2, TB, 0, stream>>>(x, clsw, clsb, skip, xxA);

  // ---- Mamba layers ----
  float* cur = xxA;
  float* nxt = xxB;
  const int nElem = MROWS * DINNER;
  for (int layer = 0; layer < 2; ++layer) {
    const float* Wi  = wiAll  + (size_t)layer * 1024 * 256;
    const float* cw  = cwAll  + (size_t)layer * 512 * 4;
    const float* cb  = cbAll  + (size_t)layer * 512;
    const float* Wx  = wxAll  + (size_t)layer * 48 * 512;
    const float* Wdt = wdtAll + (size_t)layer * 512 * 16;
    const float* bdt = bdtAll + (size_t)layer * 512;
    const float* al  = alogAll + (size_t)layer * 512 * 16;
    const float* Dp  = dpAll  + (size_t)layer * 512;
    const float* Wo  = woAll  + (size_t)layer * 256 * 512;

    gemm(cur, Wi, xz, MROWS, 1024, 256);                       // in_proj
    conv_silu_kernel<<<(nElem + TB - 1) / TB, TB, 0, stream>>>(xz, cw, cb, xi);
    gemm(xi, Wx, dbc, MROWS, 48, 512);                         // x_proj
    delta_kernel<<<(nElem + TB - 1) / TB, TB, 0, stream>>>(dbc, Wdt, bdt, dlt);
    scan_kernel<<<128, 128, 0, stream>>>(dbc, dlt, xz, xi, al, Dp, yb);
    gemm(yb, Wo, nxt, MROWS, DMODEL, 512);                     // out_proj

    float* t = cur; cur = nxt; nxt = t;
  }

  // ---- final batchnorm + tuple tail ----
  bn_stats_kernel<<<256, TB, 0, stream>>>(cur, stats);
  bn_apply_kernel<<<(MROWS * DMODEL + TB - 1) / TB, TB, 0, stream>>>(
      cur, stats, bng, bnb, (float*)d_out, MROWS * DMODEL);
  tail_kernel<<<1, 32, 0, stream>>>((float*)d_out, out_size);
}
